// NTMCell_2181843386925
// MI455X (gfx1250) — matrix operand order repaired and checked by reading, not compiled, not run on hardware
//
#include <hip/hip_runtime.h>
#include <hip/hip_bf16.h>
#include <math.h>

typedef __attribute__((ext_vector_type(16))) _Float16 v16h;
typedef __attribute__((ext_vector_type(8)))  _Float16 v8h;
typedef __attribute__((ext_vector_type(8)))  float    v8f;

#define EPSF 1e-8f

__device__ __forceinline__ float sigf(float x) { return 1.0f / (1.0f + __expf(-x)); }
__device__ __forceinline__ float softplusf(float x) {
    return (x > 20.0f) ? x : log1pf(__expf(x));
}

// ---------------------------------------------------------------------------
// Weight conversion / concatenation to f16 (done per call; tiny cost, keeps
// kernel_launch stateless & deterministic).
// ---------------------------------------------------------------------------
__global__ void prep_weights(const float* __restrict__ W_ih,  // [1024][192]
                             const float* __restrict__ W_hh,  // [1024][256]
                             const float* __restrict__ W_w,   // [198][256]
                             const float* __restrict__ W_r,   // [70][256]
                             const float* __restrict__ W_out, // [128][320]
                             const float* __restrict__ b_w,   // [198]
                             const float* __restrict__ b_r,   // [70]
                             _Float16* __restrict__ WcatH,    // [1024][448]
                             _Float16* __restrict__ WheadH,   // [272][256]
                             _Float16* __restrict__ WoutH,    // [128][320]
                             float* __restrict__ biasHead)    // [272]
{
    const int S1 = 1024 * 448, S2 = 272 * 256, S3 = 128 * 320, S4 = 272;
    int idx = blockIdx.x * blockDim.x + threadIdx.x;
    if (idx < S1) {
        int n = idx / 448, k = idx % 448;
        float v = (k < 192) ? W_ih[n * 192 + k] : W_hh[n * 256 + (k - 192)];
        WcatH[idx] = (_Float16)v;
        return;
    }
    idx -= S1;
    if (idx < S2) {
        int n = idx / 256, k = idx % 256;
        float v = (n < 198) ? W_w[n * 256 + k]
                 : (n < 268) ? W_r[(n - 198) * 256 + k] : 0.0f;
        WheadH[idx] = (_Float16)v;
        return;
    }
    idx -= S2;
    if (idx < S3) { WoutH[idx] = (_Float16)W_out[idx]; return; }
    idx -= S3;
    if (idx < S4) {
        biasHead[idx] = (idx < 198) ? b_w[idx]
                       : (idx < 268) ? b_r[idx - 198] : 0.0f;
    }
}

// Pack LSTM GEMM activations: A1[b][0:128]=x_t, [128:192]=prev_read, [192:448]=h
__global__ void pack_A1(const float* __restrict__ x_t,
                        const float* __restrict__ prev_read,
                        const float* __restrict__ h,
                        _Float16* __restrict__ A1, int B)
{
    int idx = blockIdx.x * blockDim.x + threadIdx.x;
    if (idx >= B * 448) return;
    int b = idx / 448, k = idx % 448;
    float v = (k < 128) ? x_t[b * 128 + k]
             : (k < 192) ? prev_read[b * 64 + (k - 128)]
                         : h[b * 256 + (k - 192)];
    A1[idx] = (_Float16)v;
}

// ---------------------------------------------------------------------------
// Generic WMMA GEMM: C[Brows][N] = A[Brows][K] (f16) @ W[N][K]^T (f16) + bias
// Wave computes a 16 x (16*NT) tile. K % 32 == 0, Brows % 16 == 0, N % (16*NT)==0.
// Fragment k order: contiguous k = 16*hi + s in both A and B (wave32).
// ---------------------------------------------------------------------------
template <int NT, bool SIG>
__global__ void gemm_f16_wmma(const _Float16* __restrict__ A,
                              const _Float16* __restrict__ W,
                              const float* __restrict__ bias,
                              float* __restrict__ C,
                              int Brows, int N, int K, int numWaves)
{
    int wave = (blockIdx.x * blockDim.x + threadIdx.x) >> 5;
    if (wave >= numWaves) return;          // wave-uniform: EXEC all-ones for WMMA
    int lane = threadIdx.x & 31;
    int l = lane & 15, hi = lane >> 4;

    int tilesN = N / (16 * NT);
    int tm = wave / tilesN;
    int tn = wave % tilesN;

    const _Float16* arow = A + (size_t)(tm * 16 + l) * K;

    v8f acc[NT];
    v8f zero = {0.f, 0.f, 0.f, 0.f, 0.f, 0.f, 0.f, 0.f};
#pragma unroll
    for (int t = 0; t < NT; ++t) acc[t] = zero;

    for (int k0 = 0; k0 < K; k0 += 32) {
        // A fragment: lane = row (l); halves 0..7 <- K=k0+16*hi.., 8..15 <- K=k0+16*hi+8..
        v8h alo = *(const v8h*)(arow + k0 + 16 * hi);
        v8h ahi = *(const v8h*)(arow + k0 + 16 * hi + 8);
        v16h af;
#pragma unroll
        for (int i = 0; i < 8; ++i) { af[i] = alo[i]; af[i + 8] = ahi[i]; }
#pragma unroll
        for (int t = 0; t < NT; ++t) {
            // B fragment: lane = column n; 16 contiguous K values at K=k0+16*hi
            const _Float16* brow =
                W + (size_t)(tn * 16 * NT + t * 16 + l) * K + k0 + 16 * hi;
            v16h bf = *(const v16h*)brow;
            acc[t] = __builtin_amdgcn_wmma_f32_16x16x32_f16(
                false, af, false, bf, (short)0, acc[t], false, false);
        }
    }

#pragma unroll
    for (int t = 0; t < NT; ++t) {
        int col = tn * 16 * NT + t * 16 + l;
        float bv = bias[col];
#pragma unroll
        for (int v = 0; v < 8; ++v) {
            int row = tm * 16 + v + 8 * hi;   // D layout: m = v + 8*hi, n = l
            float r = acc[t][v] + bv;
            if (SIG) r = 1.0f / (1.0f + __expf(-r));
            C[(size_t)row * N + col] = r;
        }
    }
}

// ---------------------------------------------------------------------------
// LSTM pointwise: gates -> o_t, emitted as f16 into A2 (head GEMM input) and
// A3 cols [64:320) (output GEMM input).
// ---------------------------------------------------------------------------
__global__ void lstm_pw(const float* __restrict__ z,  // [B][1024] (bias included)
                        const float* __restrict__ c,  // [B][256]
                        _Float16* __restrict__ A2,    // [B][256]
                        _Float16* __restrict__ A3,    // [B][320]
                        int B)
{
    int idx = blockIdx.x * blockDim.x + threadIdx.x;
    if (idx >= B * 256) return;
    int b = idx >> 8, j = idx & 255;
    const float* zb = z + (size_t)b * 1024;
    float i_g = sigf(zb[j]);
    float f_g = sigf(zb[256 + j]);
    float g_g = tanhf(zb[512 + j]);
    float o_g = sigf(zb[768 + j]);
    float cn  = f_g * c[idx] + i_g * g_g;
    float ot  = o_g * tanhf(cn);
    _Float16 oh = (_Float16)ot;
    A2[idx] = oh;
    A3[(size_t)b * 320 + 64 + j] = oh;
}

// ---------------------------------------------------------------------------
// Fused write-head + read-head per batch sample. mem slab lives in LDS, so
// mem_new is never written to HBM (it is not a kernel output).
// Block = 128 threads, one block per sample.
// ---------------------------------------------------------------------------
__device__ __forceinline__ float reduce128(float* s, int t, float v, bool ismax)
{
    s[t] = v;
    __syncthreads();
    for (int o = 64; o > 0; o >>= 1) {
        if (t < o) s[t] = ismax ? fmaxf(s[t], s[t + o]) : (s[t] + s[t + o]);
        __syncthreads();
    }
    float r = s[0];
    __syncthreads();
    return r;
}

__global__ void heads_kernel(const float* __restrict__ P,        // [B][272]
                             const float* __restrict__ mem,      // [B][128][64]
                             const float* __restrict__ w_write,  // [B][128]
                             const float* __restrict__ w_read,   // [B][128]
                             _Float16* __restrict__ A3)          // [B][320]: cols 0..63
{
    __shared__ float sMem[128 * 64];   // 32 KB memory slab
    __shared__ float sK[64];
    __shared__ float sE[64];
    __shared__ float sA[64];
    __shared__ float sW[128];          // current weighting
    __shared__ float sVal[128];        // softmax values
    __shared__ float sTmp[128];        // reductions
    __shared__ float sScal[7];         // beta,g,s0,s1,s2,gamma

    int b = blockIdx.x;
    int t = threadIdx.x;
    const float* pb = P + (size_t)b * 272;

    // Stage mem[b] into LDS (2048 float4 / 128 threads)
    {
        const float4* m4 = (const float4*)(mem + (size_t)b * 8192);
        float4* s4 = (float4*)sMem;
        for (int i = t; i < 2048; i += 128) s4[i] = m4[i];
    }
    __syncthreads();

    for (int pass = 0; pass < 2; ++pass) {
        int base = pass ? 198 : 0;
        const float* wprev = pass ? (w_read + (size_t)b * 128)
                                  : (w_write + (size_t)b * 128);

        if (t < 64) sK[t] = tanhf(pb[base + t]);
        if (t == 0) {
            sScal[0] = softplusf(pb[base + 64]);         // beta
            sScal[1] = sigf(pb[base + 65]);              // g
            float s0 = pb[base + 66], s1 = pb[base + 67], s2 = pb[base + 68];
            float mx = fmaxf(s0, fmaxf(s1, s2));
            float e0 = __expf(s0 - mx), e1 = __expf(s1 - mx), e2 = __expf(s2 - mx);
            float se = e0 + e1 + e2;
            sScal[2] = e0 / se; sScal[3] = e1 / se; sScal[4] = e2 / se;
            sScal[5] = 1.0f + softplusf(pb[base + 69]);  // gamma
        }
        if (pass == 0 && t < 64) {
            sE[t] = sigf(pb[70 + t]);
            sA[t] = tanhf(pb[134 + t]);
        }
        __syncthreads();

        float beta = sScal[0], g = sScal[1];
        float s0 = sScal[2], s1 = sScal[3], s2 = sScal[4], gamma = sScal[5];

        float ksum = reduce128(sTmp, t, (t < 64) ? sK[t] * sK[t] : 0.0f, false);
        float knorm = sqrtf(ksum);

        // content addressing: thread t owns row n = t (staggered to dodge banks)
        float dot = 0.0f, nrm = 0.0f;
        {
            const float* row = sMem + t * 64;
#pragma unroll 8
            for (int mm = 0; mm < 64; ++mm) {
                int m = (mm + t) & 63;
                float v = row[m];
                dot += v * sK[m];
                nrm += v * v;
            }
        }
        float sim = dot / ((sqrtf(nrm) + EPSF) * (knorm + EPSF));
        float logit = beta * sim;
        sVal[t] = logit;
        __syncthreads();

        float mx = reduce128(sTmp, t, logit, true);
        float ex = __expf(logit - mx);
        float ssum = reduce128(sTmp, t, ex, false);
        float wc = ex / ssum;

        float wg = g * wc + (1.0f - g) * wprev[t];
        sW[t] = wg;
        __syncthreads();
        float wsv = s0 * sW[(t + 127) & 127] + s1 * wg + s2 * sW[(t + 1) & 127];
        __syncthreads();
        float wp = __powf(fmaxf(wsv, EPSF), gamma);
        float psum = reduce128(sTmp, t, wp, false);
        float wwn = wp / (psum + EPSF);
        sW[t] = wwn;
        __syncthreads();

        if (pass == 0) {
            // erase/add update, in LDS only
            float4* s4 = (float4*)sMem;
            for (int i = t; i < 2048; i += 128) {
                int n = i >> 4;
                int mb = (i & 15) * 4;
                float w = sW[n];
                float4 v = s4[i];
                v.x = v.x * (1.0f - w * sE[mb + 0]) + w * sA[mb + 0];
                v.y = v.y * (1.0f - w * sE[mb + 1]) + w * sA[mb + 1];
                v.z = v.z * (1.0f - w * sE[mb + 2]) + w * sA[mb + 2];
                v.w = v.w * (1.0f - w * sE[mb + 3]) + w * sA[mb + 3];
                s4[i] = v;
            }
            __syncthreads();
        } else {
            // r_t[m] = sum_n wr[n] * mem_new[n][m]; column reads are conflict-free
            if (t < 64) {
                float acc = 0.0f;
#pragma unroll 8
                for (int n = 0; n < 128; ++n) acc += sW[n] * sMem[n * 64 + t];
                A3[(size_t)b * 320 + t] = (_Float16)acc;
            }
        }
    }
}

// ---------------------------------------------------------------------------
// Host launcher
// ---------------------------------------------------------------------------
extern "C" void kernel_launch(void* const* d_in, const int* in_sizes, int n_in,
                              void* d_out, int out_size, void* d_ws, size_t ws_size,
                              hipStream_t stream)
{
    const float* x_t       = (const float*)d_in[0];
    const float* prev_read = (const float*)d_in[1];
    const float* mem       = (const float*)d_in[2];
    const float* w_read    = (const float*)d_in[3];
    const float* w_write   = (const float*)d_in[4];
    const float* h         = (const float*)d_in[5];
    const float* c         = (const float*)d_in[6];
    const float* W_ih      = (const float*)d_in[7];
    const float* W_hh      = (const float*)d_in[8];
    const float* b_lstm    = (const float*)d_in[9];
    const float* W_r       = (const float*)d_in[10];
    const float* b_r       = (const float*)d_in[11];
    const float* W_w       = (const float*)d_in[12];
    const float* b_w       = (const float*)d_in[13];
    const float* W_out     = (const float*)d_in[14];
    const float* b_out     = (const float*)d_in[15];

    const int B = 4096;

    // workspace carve-up (256B aligned)
    char* ws = (char*)d_ws;
    size_t off = 0;
    auto carve = [&](size_t bytes) -> char* {
        char* p = ws + off;
        off = (off + bytes + 255) & ~(size_t)255;
        return p;
    };
    _Float16* WcatH  = (_Float16*)carve((size_t)1024 * 448 * 2);
    _Float16* WheadH = (_Float16*)carve((size_t)272 * 256 * 2);
    _Float16* WoutH  = (_Float16*)carve((size_t)128 * 320 * 2);
    float*    biasHd = (float*)carve((size_t)272 * 4);
    _Float16* A1H    = (_Float16*)carve((size_t)B * 448 * 2);
    float*    Z      = (float*)carve((size_t)B * 1024 * 4);
    _Float16* A2H    = (_Float16*)carve((size_t)B * 256 * 2);
    float*    Pf     = (float*)carve((size_t)B * 272 * 4);
    _Float16* A3H    = (_Float16*)carve((size_t)B * 320 * 2);

    // 1) weights -> f16 (concat + pad)
    {
        int total = 1024 * 448 + 272 * 256 + 128 * 320 + 272;
        prep_weights<<<(total + 255) / 256, 256, 0, stream>>>(
            W_ih, W_hh, W_w, W_r, W_out, b_w, b_r, WcatH, WheadH, WoutH, biasHd);
    }
    // 2) activations -> f16
    pack_A1<<<(B * 448 + 255) / 256, 256, 0, stream>>>(x_t, prev_read, h, A1H, B);

    // 3) z = A1 @ Wcat^T + b_lstm   (4096 x 1024, K=448)
    {
        int numWaves = (B / 16) * (1024 / 64);
        gemm_f16_wmma<4, false><<<(numWaves * 32 + 255) / 256, 256, 0, stream>>>(
            A1H, WcatH, b_lstm, Z, B, 1024, 448, numWaves);
    }
    // 4) gates -> o_t (f16 into A2 and A3[:,64:320))
    lstm_pw<<<(B * 256 + 255) / 256, 256, 0, stream>>>(Z, c, A2H, A3H, B);

    // 5) P = o_t @ Whead^T + biasHead   (4096 x 272, K=256)
    {
        int numWaves = (B / 16) * (272 / 16);
        gemm_f16_wmma<1, false><<<(numWaves * 32 + 255) / 256, 256, 0, stream>>>(
            A2H, WheadH, biasHd, Pf, B, 272, 256, numWaves);
    }
    // 6) fused write+read heads; r_t -> A3[:,0:64)
    heads_kernel<<<B, 128, 0, stream>>>(Pf, mem, w_write, w_read, A3H);

    // 7) out = sigmoid(A3 @ Wout^T + b_out)   (4096 x 128, K=320)
    {
        int numWaves = (B / 16) * (128 / 64);
        gemm_f16_wmma<4, true><<<(numWaves * 32 + 255) / 256, 256, 0, stream>>>(
            A3H, WoutH, b_out, (float*)d_out, B, 128, 320, numWaves);
    }
}
